// GeometricRelationalGraphConv_30588757082311
// MI455X (gfx1250) — matrix operand order repaired and checked by reading, not compiled, not run on hardware
//
#include <hip/hip_runtime.h>
#include <hip/hip_bf16.h>

// ---------------------------------------------------------------------------
// GeometricRelationalGraphConv forward for MI455X (gfx1250, wave32, WMMA).
// N=50000 nodes, E=800000 edges, D=128, DE=64, R=7.
// Memory-bound pipeline: bf16 WMMA GEMMs, fused BN+LReLU epilogues,
// fused gather+scatter-add with f32 global atomics (lands in 192MB L2).
// ---------------------------------------------------------------------------

typedef __bf16 bf16;
typedef __attribute__((ext_vector_type(16))) __bf16 bf16x16;
typedef __attribute__((ext_vector_type(8)))  __bf16 bf16x8;
typedef __attribute__((ext_vector_type(8)))  float  f32x8;
typedef __attribute__((ext_vector_type(4)))  float  f32x4;

#define LREL(y) ((y) >= 0.0f ? (y) : 0.1f * (y))
#define BN_EPS 1e-5f

// ---------------------------------------------------------------------------
// zero fill (float4 grid-stride)
// ---------------------------------------------------------------------------
__global__ __launch_bounds__(256) void zero_f32(float* __restrict__ p, size_t n) {
    size_t i = (size_t)blockIdx.x * blockDim.x + threadIdx.x;
    size_t stride = (size_t)gridDim.x * blockDim.x;
    size_t n4 = n >> 2;
    f32x4* p4 = (f32x4*)p;
    for (size_t j = i; j < n4; j += stride) p4[j] = (f32x4){0.f, 0.f, 0.f, 0.f};
    for (size_t j = (n4 << 2) + i; j < n; j += stride) p[j] = 0.f;
}

// ---------------------------------------------------------------------------
// weight convert+transpose: W[K,N] f32 -> Wt[N,K] bf16
// ---------------------------------------------------------------------------
__global__ __launch_bounds__(256) void wtrans_bf16(const float* __restrict__ W,
                                                   bf16* __restrict__ Wt, int K, int N) {
    int i = blockIdx.x * blockDim.x + threadIdx.x;
    if (i < K * N) {
        int k = i / N, n = i - k * N;
        Wt[(size_t)n * K + k] = (bf16)W[i];
    }
}

// ---------------------------------------------------------------------------
// BatchNorm stats: per-column sum / sum-of-squares of x[rows,cols], cols<=1024
// ---------------------------------------------------------------------------
__global__ __launch_bounds__(256) void bn_stats(const float* __restrict__ x, int rows, int cols,
                                                float* __restrict__ sum, float* __restrict__ sq) {
    float s[4] = {0.f, 0.f, 0.f, 0.f};
    float q[4] = {0.f, 0.f, 0.f, 0.f};
    const int t = threadIdx.x;
    for (int r = blockIdx.x; r < rows; r += gridDim.x) {
        const float* row = x + (size_t)r * cols;
#pragma unroll
        for (int i = 0; i < 4; ++i) {
            int c = t + (i << 8);
            if (c < cols) { float v = row[c]; s[i] += v; q[i] += v * v; }
        }
    }
#pragma unroll
    for (int i = 0; i < 4; ++i) {
        int c = t + (i << 8);
        if (c < cols) { atomicAdd(&sum[c], s[i]); atomicAdd(&sq[c], q[i]); }
    }
}

// ---------------------------------------------------------------------------
// BN (training) + LeakyReLU apply, f32 or bf16 output
// ---------------------------------------------------------------------------
__device__ __forceinline__ float bn_lrelu_elem(float v, int c, const float* sum, const float* sq,
                                               const float* g, const float* b, float invR) {
    float m  = sum[c] * invR;
    float va = sq[c] * invR - m * m;
    float sc = g[c] * rsqrtf(va + BN_EPS);
    float y  = (v - m) * sc + b[c];
    return LREL(y);
}

__global__ __launch_bounds__(256) void bn_lrelu_to_bf16(const float* __restrict__ x, size_t total,
        int cols, const float* __restrict__ sum, const float* __restrict__ sq,
        const float* __restrict__ g, const float* __restrict__ b, float invR,
        bf16* __restrict__ out) {
    size_t stride = (size_t)gridDim.x * blockDim.x;
    for (size_t i = (size_t)blockIdx.x * blockDim.x + threadIdx.x; i < total; i += stride) {
        int c = (int)(i % (size_t)cols);
        out[i] = (bf16)bn_lrelu_elem(x[i], c, sum, sq, g, b, invR);
    }
}

__global__ __launch_bounds__(256) void bn_lrelu_to_f32(const float* __restrict__ x, size_t total,
        int cols, const float* __restrict__ sum, const float* __restrict__ sq,
        const float* __restrict__ g, const float* __restrict__ b, float invR,
        float* __restrict__ out) {
    size_t stride = (size_t)gridDim.x * blockDim.x;
    for (size_t i = (size_t)blockIdx.x * blockDim.x + threadIdx.x; i < total; i += stride) {
        int c = (int)(i % (size_t)cols);
        out[i] = bn_lrelu_elem(x[i], c, sum, sq, g, b, invR);
    }
}

// ---------------------------------------------------------------------------
// GEMM: C[M,Ncols] = A[M,K](bf16,row-major) x Bt[Ncols,K](bf16,row-major,
// i.e. B transposed) [+ bias[Ncols]] [+ resid[M,Ncols]], f32 accumulate.
// Block = 256 threads = 8 waves; wave w owns the 16x16 tile at cols [16w,16w+16).
// Grid.x = ceil(M/16). Uses v_wmma_f32_16x16x32_bf16.
//
// Fragment layouts per CDNA5 ISA 7.12.2 (16-bit, wave32):
//  A: lane holds row m=lane&15; elems 0..7 = K {half*8..+7}, 8..15 = K {16+half*8..+7}
//  B: lane holds col n=lane&15; elems 0..7 = K {half*8..+7}, 8..15 = K {16+half*8..+7}
//  C: lane col n=lane&15; elem j -> row rowBase + j + half*8
// ---------------------------------------------------------------------------
__global__ __launch_bounds__(256) void gemm_bf16_wmma(
        const bf16* __restrict__ A, const bf16* __restrict__ Bt,
        const float* __restrict__ bias, const float* __restrict__ resid,
        float* __restrict__ C, int M, int K, int Ncols) {
    const int wave = threadIdx.x >> 5;
    const int lane = threadIdx.x & 31;
    const int half = lane >> 4;
    const int l16  = lane & 15;
    const int rowBase = blockIdx.x << 4;
    const int colBase = wave << 4;
    if (colBase >= Ncols) return;

    int m = rowBase + l16;
    if (m >= M) m = M - 1;                      // clamp (EXEC must stay full for WMMA)
    const int n = colBase + l16;

    const bf16* __restrict__ arow = A  + (size_t)m * K;
    const bf16* __restrict__ brow = Bt + (size_t)n * K;

    f32x8 acc = {};
    for (int k = 0; k < K; k += 32) {
        bf16x8 a0 = *reinterpret_cast<const bf16x8*>(arow + k + half * 8);
        bf16x8 a1 = *reinterpret_cast<const bf16x8*>(arow + k + 16 + half * 8);
        bf16x8 b0 = *reinterpret_cast<const bf16x8*>(brow + k + half * 8);
        bf16x8 b1 = *reinterpret_cast<const bf16x8*>(brow + k + 16 + half * 8);
        bf16x16 bfr;
        bf16x16 afr;
#pragma unroll
        for (int j = 0; j < 8; ++j) { afr[j] = a0[j]; afr[j + 8] = a1[j]; bfr[j] = b0[j]; bfr[j + 8] = b1[j]; }
        acc = __builtin_amdgcn_wmma_f32_16x16x32_bf16(
                  false, afr, false, bfr, (short)0, acc, false, false);
    }

    const float bval = bias ? bias[n] : 0.0f;
#pragma unroll
    for (int j = 0; j < 8; ++j) {
        int r = rowBase + half * 8 + j;
        if (r < M) {
            size_t idx = (size_t)r * Ncols + n;
            float v = acc[j] + bval;
            if (resid) v += resid[idx];
            C[idx] = v;
        }
    }
}

// ---------------------------------------------------------------------------
// Fused: e2 = LReLU(BN(ey)); msg = e2 + x2[node_in]; atomic scatter-add into
// update[(node_out*R + rel)*128 + c].  2 edges per 256-thread block.
// ---------------------------------------------------------------------------
__global__ __launch_bounds__(256) void edge_bn_gather_scatter(
        const float* __restrict__ ey, const float* __restrict__ sum, const float* __restrict__ sq,
        const float* __restrict__ g, const float* __restrict__ b, float invE,
        const float* __restrict__ x2, const int* __restrict__ nin,
        const int* __restrict__ nout, const int* __restrict__ rel,
        float* __restrict__ update, int E) {
    const int e = blockIdx.x * 2 + (threadIdx.x >> 7);
    const int c = threadIdx.x & 127;
    if (e >= E) return;
    float y = bn_lrelu_elem(ey[(size_t)e * 128 + c], c, sum, sq, g, b, invE);
    y += x2[(size_t)nin[e] * 128 + c];
    atomicAdd(&update[((size_t)nout[e] * 7 + (size_t)rel[e]) * 128 + c], y);
}

// ---------------------------------------------------------------------------
// host launch
// ---------------------------------------------------------------------------
static inline size_t align256(size_t x) { return (x + 255) & ~(size_t)255; }

extern "C" void kernel_launch(void* const* d_in, const int* in_sizes, int n_in,
                              void* d_out, int out_size, void* d_ws, size_t ws_size,
                              hipStream_t stream) {
    (void)n_in; (void)out_size; (void)ws_size;
    const int D = 128, DE = 64, R = 7, RD = R * D;      // 896
    const int N = in_sizes[0] / D;
    const int E = in_sizes[1];

    const float* h_v      = (const float*)d_in[0];
    const int*   node_in  = (const int*)  d_in[1];
    const int*   node_out = (const int*)  d_in[2];
    const int*   rel      = (const int*)  d_in[3];
    const float* h_e      = (const float*)d_in[4];
    const float* in_bn1_g = (const float*)d_in[5];
    const float* in_bn1_b = (const float*)d_in[6];
    const float* in_W     = (const float*)d_in[7];
    const float* in_b     = (const float*)d_in[8];
    const float* in_bn2_g = (const float*)d_in[9];
    const float* in_bn2_b = (const float*)d_in[10];
    const float* e_bn1_g  = (const float*)d_in[11];
    const float* e_bn1_b  = (const float*)d_in[12];
    const float* e_W      = (const float*)d_in[13];
    const float* e_b      = (const float*)d_in[14];
    const float* e_bn2_g  = (const float*)d_in[15];
    const float* e_bn2_b  = (const float*)d_in[16];
    const float* lin_bn_g = (const float*)d_in[17];
    const float* lin_bn_b = (const float*)d_in[18];
    const float* lin_W    = (const float*)d_in[19];
    const float* out_bn_g = (const float*)d_in[20];
    const float* out_bn_b = (const float*)d_in[21];
    const float* out_W    = (const float*)d_in[22];
    float* out = (float*)d_out;

    // -------- workspace arena (lifetime-aliased) --------
    char* ws = (char*)d_ws;
    const size_t base1 = align256((size_t)N * D * 4);           // after X2
    const size_t base2 = base1 + align256((size_t)E * D * 4);   // after EY
    const size_t base3 = base2 + align256((size_t)N * RD * 4);  // after UPDATE

    float* X2     = (float*)(ws);                                // [N,128] f32
    bf16*  XH     = (bf16*)(ws + base1);                         // [N,128] bf16 (early)
    float* Y1     = (float*)(ws + base1 + align256((size_t)N * D * 2));
    float* EY     = (float*)(ws + base1);                        // [E,128] f32 (mid)
    bf16*  EHAT   = (bf16*)(ws + base2);                         // [E,64] bf16 (early)
    float* UPDATE = (float*)(ws + base2);                        // [N*R,128] f32 (late)
    bf16*  UHAT   = (bf16*)(ws + base1);                         // [N,896] bf16 (late)
    float* O1     = (float*)(ws + base1 + align256((size_t)N * RD * 2));
    bf16*  O1H    = (bf16*)(ws + base1 + align256((size_t)N * RD * 2) + align256((size_t)N * D * 4));
    bf16*  inWt   = (bf16*)(ws + base3);                         // [128,128]
    bf16*  eWt    = inWt + D * D;                                // [128,64]
    bf16*  linWt  = eWt + DE * D;                                // [128,896]
    bf16*  outWt  = linWt + RD * D;                              // [128,128]
    float* SSUM   = (float*)(ws + base3 + align256(((size_t)(D*D + DE*D + RD*D + D*D)) * 2));
    float* SSQ    = SSUM + 1024;

    auto gridFor = [](size_t total) -> unsigned {
        size_t bl = (total + 255) / 256;
        return (unsigned)(bl > 262144 ? 262144 : bl);
    };
    auto stats = [&](const float* x, int rows, int cols) {
        zero_f32<<<8, 256, 0, stream>>>(SSUM, 2048);
        int blocks = rows < 2048 ? rows : 2048;
        bn_stats<<<blocks, 256, 0, stream>>>(x, rows, cols, SSUM, SSQ);
    };

    // -------- weight prep (bf16, transposed) --------
    wtrans_bf16<<<(D * D + 255) / 256, 256, 0, stream>>>(in_W, inWt, D, D);
    wtrans_bf16<<<(DE * D + 255) / 256, 256, 0, stream>>>(e_W, eWt, DE, D);
    wtrans_bf16<<<(RD * D + 255) / 256, 256, 0, stream>>>(lin_W, linWt, RD, D);
    wtrans_bf16<<<(D * D + 255) / 256, 256, 0, stream>>>(out_W, outWt, D, D);

    // -------- node input MLP --------
    stats(h_v, N, D);
    bn_lrelu_to_bf16<<<gridFor((size_t)N * D), 256, 0, stream>>>(
        h_v, (size_t)N * D, D, SSUM, SSQ, in_bn1_g, in_bn1_b, 1.0f / N, XH);
    gemm_bf16_wmma<<<(N + 15) / 16, 256, 0, stream>>>(XH, inWt, in_b, nullptr, Y1, N, D, D);
    stats(Y1, N, D);
    bn_lrelu_to_f32<<<gridFor((size_t)N * D), 256, 0, stream>>>(
        Y1, (size_t)N * D, D, SSUM, SSQ, in_bn2_g, in_bn2_b, 1.0f / N, X2);

    // -------- edge MLP --------
    stats(h_e, E, DE);
    bn_lrelu_to_bf16<<<gridFor((size_t)E * DE), 256, 0, stream>>>(
        h_e, (size_t)E * DE, DE, SSUM, SSQ, e_bn1_g, e_bn1_b, 1.0f / E, EHAT);
    gemm_bf16_wmma<<<(E + 15) / 16, 256, 0, stream>>>(EHAT, eWt, e_b, nullptr, EY, E, DE, D);
    stats(EY, E, D);

    // -------- zero accumulator, fused BN+LReLU+gather+scatter --------
    zero_f32<<<gridFor((size_t)N * RD / 4), 256, 0, stream>>>(UPDATE, (size_t)N * RD);
    edge_bn_gather_scatter<<<(E + 1) / 2, 256, 0, stream>>>(
        EY, SSUM, SSQ, e_bn2_g, e_bn2_b, 1.0f / E, X2, node_in, node_out, rel, UPDATE, E);

    // -------- linear over [N, R*D] --------
    stats(UPDATE, N, RD);
    bn_lrelu_to_bf16<<<gridFor((size_t)N * RD), 256, 0, stream>>>(
        UPDATE, (size_t)N * RD, RD, SSUM, SSQ, lin_bn_g, lin_bn_b, 1.0f / N, UHAT);
    gemm_bf16_wmma<<<(N + 15) / 16, 256, 0, stream>>>(UHAT, linWt, nullptr, nullptr, O1, N, RD, D);

    // -------- output + residual --------
    stats(O1, N, D);
    bn_lrelu_to_bf16<<<gridFor((size_t)N * D), 256, 0, stream>>>(
        O1, (size_t)N * D, D, SSUM, SSQ, out_bn_g, out_bn_b, 1.0f / N, O1H);
    gemm_bf16_wmma<<<(N + 15) / 16, 256, 0, stream>>>(O1H, outWt, nullptr, h_v, out, N, D, D);
}
